// ReluModular_68805376081922
// MI455X (gfx1250) — matrix operand order repaired and checked by reading, not compiled, not run on hardware
//
#include <hip/hip_runtime.h>

// ---------------- types -----------------
typedef _Float16 h2f  __attribute__((ext_vector_type(2)));
typedef _Float16 h8f  __attribute__((ext_vector_type(8)));
typedef _Float16 h16f __attribute__((ext_vector_type(16)));
typedef float    f8v  __attribute__((ext_vector_type(8)));

union H16 { h16f v; h8f h[2]; };

// ---------------- fast activations (hardware exp2/rcp) -----------------
__device__ __forceinline__ float fsig(float x) {
  return __builtin_amdgcn_rcpf(1.0f + __builtin_amdgcn_exp2f(-1.44269504f * x));
}
__device__ __forceinline__ float ftanh(float x) {
  return 2.0f * __builtin_amdgcn_rcpf(1.0f + __builtin_amdgcn_exp2f(-2.88539008f * x)) - 1.0f;
}

__device__ __forceinline__ f8v wmma16(h16f a, h16f b, f8v c) {
  return __builtin_amdgcn_wmma_f32_16x16x32_f16(false, a, false, b, (short)0, c, false, false);
}

// ---------------- problem constants -----------------
#define BB   4096
#define SEQ  512
#define EE   20
#define H1   64
#define H2   256

#define XT_S 32   // per-tile x-embed LDS row stride (K padded 20->32), halves
#define HT_S 64   // per-tile h LDS row stride (K = 64), halves

// =====================================================================
// Stage 1: embedding + 512-step LSTM (H1=64).
// One block (4 waves) per 16 batch rows; wave w owns hidden-tile w
// (gate N-tiles w, 4+w, 8+w, 12+w). All 12 B-fragments (weights) stay
// resident in VGPRs for the whole scan; h/x exchanged through ping-pong
// LDS tiles with a single s_barrier per timestep.
// =====================================================================
__global__ __launch_bounds__(128) void lstm1_kernel(
    const float* __restrict__ x,        // (4096, 1024)
    const float* __restrict__ W_embed,  // (20, 2)
    const float* __restrict__ Wih1,     // (256, 20)
    const float* __restrict__ Whh1,     // (256, 64)
    const float* __restrict__ b1,       // (256,)
    _Float16* __restrict__ h_out)       // (4096, 64) f16 workspace
{
  __shared__ _Float16 xtile[2][16 * XT_S];   // ping-pong token embeds (2 KB)
  __shared__ _Float16 htile[2][16 * HT_S];   // ping-pong hidden state (4 KB)

  const int tid  = threadIdx.x;
  const int wave = tid >> 5;                 // 0..3  (= hidden-tile owned)
  const int lane = tid & 31;
  const int lrow = lane & 15;
  const int hi   = lane >> 4;                // 0: lanes 0-15, 1: lanes 16-31
  const int aoff = hi * 16;                  // A-frag half offset
  const int boff = hi * 16;                  // B-frag K offset

  const int batch0 = blockIdx.x * 16;        // 256 blocks x 16 rows = 4096
  const long growA = batch0 + lrow;

  // ---- zero h(−1) and x pad columns ----
  for (int i = tid; i < 16 * HT_S; i += 128) htile[0][i] = (_Float16)0.0f;
  for (int i = tid; i < 16 * XT_S; i += 128) {
    xtile[0][i] = (_Float16)0.0f;
    xtile[1][i] = (_Float16)0.0f;
  }

  // ---- load this wave's 12 B-fragments into registers (stay resident) ----
  H16 bw[4];        // Wih1.T fragments, K=20 padded to 32
  H16 bh[4][2];     // Whh1.T fragments, K=64 (two K=32 chunks)
  float biasv[4];
#pragma unroll
  for (int gt = 0; gt < 4; ++gt) {
    const int n = (gt * 4 + wave) * 16 + lrow;   // gate output column
    biasv[gt] = b1[n];
#pragma unroll
    for (int j = 0; j < 16; ++j) {
      const int k = boff + j;
      bw[gt].v[j] = (k < EE) ? (_Float16)Wih1[n * EE + k] : (_Float16)0.0f;
    }
#pragma unroll
    for (int kc = 0; kc < 2; ++kc) {
      const float4* p = (const float4*)(Whh1 + (long)n * H1 + kc * 32 + boff);
#pragma unroll
      for (int q = 0; q < 4; ++q) {
        float4 f = p[q];
        bh[gt][kc].v[4 * q + 0] = (_Float16)f.x;
        bh[gt][kc].v[4 * q + 1] = (_Float16)f.y;
        bh[gt][kc].v[4 * q + 2] = (_Float16)f.z;
        bh[gt][kc].v[4 * q + 3] = (_Float16)f.w;
      }
    }
  }

  // ---- wave 0: embed weights (lane computes k = klo..klo+9 of its row) ----
  const int klo = hi ? 10 : 0;
  float we0[10], we1[10];
  if (wave == 0) {
#pragma unroll
    for (int j = 0; j < 10; ++j) {
      we0[j] = W_embed[(klo + j) * 2 + 0];
      we1[j] = W_embed[(klo + j) * 2 + 1];
    }
    // prologue: embed token 0 into xtile[0]
    const float2 xv = *(const float2*)(x + growA * 1024);
#pragma unroll
    for (int j = 0; j < 10; j += 2) {
      float e0 = fmaxf(xv.x * we0[j]     + xv.y * we1[j],     0.0f);
      float e1 = fmaxf(xv.x * we0[j + 1] + xv.y * we1[j + 1], 0.0f);
      h2f p; p.x = (_Float16)e0; p.y = (_Float16)e1;
      *(h2f*)&xtile[0][lrow * XT_S + klo + j] = p;
    }
  }
  __syncthreads();

  f8v c = {};                                 // this wave's 16x16 cell slice

  for (int t = 0; t < SEQ; ++t) {
    const int rb = t & 1;                     // read buffer (h(t-1), x(t))
    const int wb = rb ^ 1;                    // write buffer (h(t), x(t+1))

    // ---- wave 0 pipelines next token's embedding into xtile[wb] ----
    if (wave == 0 && t + 1 < SEQ) {
      const float2 xv = *(const float2*)(x + growA * 1024 + 2 * (t + 1));
      if (t + 32 < SEQ) __builtin_prefetch(x + growA * 1024 + 2 * (t + 32), 0, 3);
#pragma unroll
      for (int j = 0; j < 10; j += 2) {
        float e0 = fmaxf(xv.x * we0[j]     + xv.y * we1[j],     0.0f);
        float e1 = fmaxf(xv.x * we0[j + 1] + xv.y * we1[j + 1], 0.0f);
        h2f p; p.x = (_Float16)e0; p.y = (_Float16)e1;
        *(h2f*)&xtile[wb][lrow * XT_S + klo + j] = p;
      }
    }

    // ---- A fragments (16-bit A layout; two b128 loads each) ----
    const _Float16* xt = &xtile[rb][0];
    const _Float16* ht = &htile[rb][0];
    H16 ax, ah0, ah1;
    ax.h[0]  = *(const h8f*)&xt[lrow * XT_S + aoff];
    ax.h[1]  = *(const h8f*)&xt[lrow * XT_S + 8 + aoff];
    ah0.h[0] = *(const h8f*)&ht[lrow * HT_S + aoff];
    ah0.h[1] = *(const h8f*)&ht[lrow * HT_S + 8 + aoff];
    ah1.h[0] = *(const h8f*)&ht[lrow * HT_S + 32 + aoff];
    ah1.h[1] = *(const h8f*)&ht[lrow * HT_S + 40 + aoff];

    // ---- gates for this wave's hidden-tile: i, f, g, o ----
    f8v g[4];
#pragma unroll
    for (int gt = 0; gt < 4; ++gt) {
      const float bv = biasv[gt];
      f8v acc = { bv, bv, bv, bv, bv, bv, bv, bv };
      acc = wmma16(ax.v,  bw[gt].v,    acc);
      acc = wmma16(ah0.v, bh[gt][0].v, acc);
      acc = wmma16(ah1.v, bh[gt][1].v, acc);
      g[gt] = acc;
    }

    // ---- c' = sig(f)*c + sig(i)*tanh(g); h' = sig(o)*tanh(c') ----
#pragma unroll
    for (int r = 0; r < 8; ++r) {
      float ig = fsig(g[0][r]);
      float fg = fsig(g[1][r]);
      float gg = ftanh(g[2][r]);
      float og = fsig(g[3][r]);
      float cn = fg * c[r] + ig * gg;
      c[r] = cn;
      float hn = og * ftanh(cn);
      // C-layout position: row m = r+8*hi, col = wave*16 + lrow
      htile[wb][(r + 8 * hi) * HT_S + wave * 16 + lrow] = (_Float16)hn;
    }

    __syncthreads();   // h(t) + x(t+1) visible; also guards WAR on read bufs
  }

  // final h(511) lives in htile[0]  (write buf of t=511 is (511&1)^1 = 0)
  for (int i = tid; i < 16 * H1; i += 128) {
    int m = i >> 6, k = i & 63;
    h_out[(long)(batch0 + m) * H1 + k] = htile[0][m * HT_S + k];
  }
}

// =====================================================================
// Stage 2: LSTM-2 single step (h=c=0 => f-gate dead) + out projection
// =====================================================================
__device__ __forceinline__ h16f load_bfrag_f32(const float* wrow) {
  h16f b{};
  const float4* p = (const float4*)wrow;
#pragma unroll
  for (int q = 0; q < 4; ++q) {
    float4 f = p[q];
    b[4 * q + 0] = (_Float16)f.x;
    b[4 * q + 1] = (_Float16)f.y;
    b[4 * q + 2] = (_Float16)f.z;
    b[4 * q + 3] = (_Float16)f.w;
  }
  return b;
}

__global__ __launch_bounds__(256) void lstm2_kernel(
    const _Float16* __restrict__ h64,   // (4096, 64) f16
    const float* __restrict__ Wih2,     // (1024, 64)
    const float* __restrict__ b2,       // (1024,)
    const float* __restrict__ Wout,     // (2, 256)
    const float* __restrict__ bout,     // (2,)
    float* __restrict__ out)            // (4096, 2)
{
  const int tid  = threadIdx.x;
  const int wave = tid >> 5;
  const int lane = tid & 31;
  const int lrow = lane & 15;
  const int hi   = lane >> 4;
  const int m0   = blockIdx.x * 128 + wave * 16;
  const long row = m0 + lrow;
  const int aoff = hi * 16;
  const int boff = hi * 16;

  // A fragments of h64 (16x64 f16, row-major in global)
  const _Float16* hrow = h64 + row * H1;
  H16 a0, a1;
  a0.h[0] = *(const h8f*)(hrow + aoff);
  a0.h[1] = *(const h8f*)(hrow + 8 + aoff);
  a1.h[0] = *(const h8f*)(hrow + 32 + aoff);
  a1.h[1] = *(const h8f*)(hrow + 40 + aoff);

  float po0[8] = {0.0f}, po1[8] = {0.0f};  // partial out for this lane's 8 rows

  for (int htl = 0; htl < 16; ++htl) {     // 16 hidden-tiles of 16 units
    const int col = htl * 16 + lrow;       // hidden unit 0..255
    static const int gsel[3] = {0, 2, 3};  // i, g, o  (f dead since c=0)
    f8v g[3];
#pragma unroll
    for (int q = 0; q < 3; ++q) {
      const int n  = gsel[q] * H2 + col;   // gate row in Wih2 / b2
      const float bv = b2[n];
      f8v acc = { bv, bv, bv, bv, bv, bv, bv, bv };
      const float* wrow = Wih2 + (long)n * H1;
      acc = wmma16(a0.v, load_bfrag_f32(wrow + boff), acc);
      acc = wmma16(a1.v, load_bfrag_f32(wrow + 32 + boff), acc);
      g[q] = acc;
    }
    const float w0 = Wout[col];
    const float w1 = Wout[H2 + col];
#pragma unroll
    for (int r = 0; r < 8; ++r) {
      float cn = fsig(g[0][r]) * ftanh(g[1][r]);   // c' = sig(i)*tanh(g)
      float hv = fsig(g[2][r]) * ftanh(cn);        // h  = sig(o)*tanh(c')
      po0[r] += hv * w0;
      po1[r] += hv * w1;
    }
  }

  // reduce over the 16 columns held by each lane group
#pragma unroll
  for (int r = 0; r < 8; ++r) {
#pragma unroll
    for (int s = 8; s >= 1; s >>= 1) {
      po0[r] += __shfl_xor(po0[r], s, 16);
      po1[r] += __shfl_xor(po1[r], s, 16);
    }
  }
  if (lrow == 0) {
    const float b0 = bout[0], b1o = bout[1];
#pragma unroll
    for (int r = 0; r < 8; ++r) {
      int m = m0 + r + 8 * hi;
      out[m * 2 + 0] = po0[r] + b0;
      out[m * 2 + 1] = po1[r] + b1o;
    }
  }
}

// =====================================================================
extern "C" void kernel_launch(void* const* d_in, const int* in_sizes, int n_in,
                              void* d_out, int out_size, void* d_ws, size_t ws_size,
                              hipStream_t stream) {
  const float* x       = (const float*)d_in[0];
  const float* W_embed = (const float*)d_in[1];
  const float* Wih1    = (const float*)d_in[2];
  const float* Whh1    = (const float*)d_in[3];
  const float* b1      = (const float*)d_in[4];
  const float* Wih2    = (const float*)d_in[5];
  // d_in[6] = Whh2 (unused: h=0 for the single LSTM-2 step)
  const float* b2      = (const float*)d_in[7];
  const float* Wout    = (const float*)d_in[8];
  const float* bout    = (const float*)d_in[9];

  _Float16* h64 = (_Float16*)d_ws;          // 4096*64 f16 = 512 KB scratch

  // 256 blocks x 4 waves: one block per 16-row batch tile, 1024 waves total
  lstm1_kernel<<<256, 128, 0, stream>>>(x, W_embed, Wih1, Whh1, b1, h64);
  // 32 blocks x 256 threads = 256 waves, each one 16-row M tile
  lstm2_kernel<<<32, 256, 0, stream>>>(h64, Wih2, b2, Wout, bout, (float*)d_out);
}
